// AttentionHead_48223892799966
// MI455X (gfx1250) — matrix operand order repaired and checked by reading, not compiled, not run on hardware
//
#include <hip/hip_runtime.h>

// ---------------------------------------------------------------------------
// Causal attention head, S=8192, d_model=d_k=d_v=64, fp32 in/out.
// Kernel 1: QKV projection (fp32 VALU) -> bf16 Q (pre-scaled by log2e/8),
//           bf16 K (row-major), bf16 V^T (transposed) in d_ws.
// Kernel 2: flash attention, TRANSPOSED formulation:
//             S^T = K * Q^T   (per-lane softmax stats, no butterfly)
//             O^T = V^T * P^T (per-lane rescale, vectorized epilogue)
//           v_wmma_f32_16x16x32_bf16 with f32 accumulators; per-wave LDS
//           bounce (4x ds_store_b128 + 4x ds_load_b128) for the P relayout.
// ---------------------------------------------------------------------------

typedef __attribute__((ext_vector_type(16))) __bf16 v16bf;
typedef __attribute__((ext_vector_type(8)))  __bf16 v8bf;
typedef __attribute__((ext_vector_type(8)))  float  v8f;
typedef __attribute__((ext_vector_type(4)))  float  v4f;

#define SEQ 8192
#define DM  64

union F16 { v16bf v; v8bf h[2]; };
union F8F { v8f v; v4f q[2]; };

// ---------------------------------------------------------------------------
// Kernel 1: q = (x @ Wq^T + bq) * (log2e/8) ; k = x @ Wk^T + bk ;
//           vT = (x @ Wv^T + bv)^T.   256 threads = 4 rows x 64 cols.
// ---------------------------------------------------------------------------
__global__ __launch_bounds__(256) void qkv_project(
    const float* __restrict__ x,
    const float* __restrict__ wq, const float* __restrict__ bq,
    const float* __restrict__ wk, const float* __restrict__ bk,
    const float* __restrict__ wv, const float* __restrict__ bv,
    __bf16* __restrict__ q, __bf16* __restrict__ k, __bf16* __restrict__ vT)
{
  __shared__ float xs[4][DM];
  const int lrow = threadIdx.x >> 6;       // 0..3
  const int c    = threadIdx.x & 63;       // 0..63 (output feature)
  const int row  = blockIdx.x * 4 + lrow;  // 0..8191

  xs[lrow][c] = x[(size_t)row * DM + c];
  __syncthreads();

  const float* xr  = xs[lrow];
  const float* wqr = wq + c * DM;
  const float* wkr = wk + c * DM;
  const float* wvr = wv + c * DM;
  float aq = bq[c], ak = bk[c], av = bv[c];
#pragma unroll 8
  for (int i = 0; i < DM; ++i) {
    const float xi = xr[i];
    aq = fmaf(xi, wqr[i], aq);
    ak = fmaf(xi, wkr[i], ak);
    av = fmaf(xi, wvr[i], av);
  }
  // fold 1/sqrt(d_k) * log2(e) so softmax can use native exp2
  q[(size_t)row * DM + c]   = (__bf16)(aq * (0.125f * 1.44269504088896340736f));
  k[(size_t)row * DM + c]   = (__bf16)ak;
  vT[(size_t)c * SEQ + row] = (__bf16)av;  // transposed: contiguous keys per d
}

// ---------------------------------------------------------------------------
// Kernel 2: transposed flash attention. 8 waves/block, 16 queries per wave
// (one query per lane-column), 64-key blocks, stop at the causal diagonal.
// ---------------------------------------------------------------------------
__global__ __launch_bounds__(256) void flash_attn(
    const __bf16* __restrict__ q, const __bf16* __restrict__ k,
    const __bf16* __restrict__ vT, float* __restrict__ out)
{
  // per-wave private P^T buffer: [query 0..15][key 0..63] bf16 (2KB/wave)
  __shared__ __attribute__((aligned(16))) __bf16 pbuf[8][16][64];

  const int wave  = threadIdx.x >> 5;
  const int lane  = threadIdx.x & 31;
  const int n     = lane & 15;   // this lane's query column / A-row index
  const int h     = lane >> 4;   // lane half
  const int qbase = blockIdx.x * 128 + wave * 16;
  const int qrow  = qbase + n;   // the query this lane owns stats for

  // --- Q as B-fragments (B K-map: (j%8) + 8h + 16(j/8)): two 16B chunks per lane
  F16 bq_[2];
  {
    const __bf16* qp = q + (size_t)qrow * DM;
#pragma unroll
    for (int kk = 0; kk < 2; ++kk) {
      bq_[kk].h[0] = *(const v8bf*)(qp + 32 * kk + 8 * h);
      bq_[kk].h[1] = *(const v8bf*)(qp + 32 * kk + 16 + 8 * h);
    }
  }

  v8f   o[4] = {};                 // O^T accumulator: (d = 16*td + r + 8h, query n)
  float mstate = -3.0e38f;         // per-lane running max (query n)
  float lstate = 0.0f;             // per-lane running sum

  const int kend = qbase + 15;
  for (int kb = 0; kb <= kend; kb += 64) {
    if (kb + 64 <= kend) {         // prefetch next key block (global_prefetch_b8)
      __builtin_prefetch(k  + (size_t)(kb + 64 + n) * DM, 0, 1);
      __builtin_prefetch(vT + (size_t)n * SEQ + kb + 64, 0, 1);
    }

    // ---- S^T = K @ Q^T : 4 key subtiles (A) x 2 d-slabs -> 8 WMMA
    // C-layout: element (key = kb + 16t + r + 8h, query n)
    v8f st[4] = {};
#pragma unroll
    for (int t = 0; t < 4; ++t) {
      const __bf16* krow = k + (size_t)(kb + 16 * t + n) * DM;
#pragma unroll
      for (int kk = 0; kk < 2; ++kk) {
        F16 ak_;   // A K-map: (j%8) + 8h + 16(j/8)
        ak_.h[0] = *(const v8bf*)(krow + 32 * kk + 8 * h);
        ak_.h[1] = *(const v8bf*)(krow + 32 * kk + 16 + 8 * h);
        st[t] = __builtin_amdgcn_wmma_f32_16x16x32_bf16(
            false, ak_.v, false, bq_[kk].v, (short)0, st[t], false, false);
      }
    }

    // ---- causal mask (diagonal-straddling blocks only), all in-lane
    if (kb + 63 > qbase) {
#pragma unroll
      for (int t = 0; t < 4; ++t)
#pragma unroll
        for (int r = 0; r < 8; ++r)
          if (kb + 16 * t + 8 * h + r > qrow) st[t][r] = -3.0e38f;
    }

    // ---- online softmax: all stats per-lane; one cross-half shuffle each
    v8f mv = __builtin_elementwise_max(__builtin_elementwise_max(st[0], st[1]),
                                       __builtin_elementwise_max(st[2], st[3]));
    float rm = mv[0];
#pragma unroll
    for (int j = 1; j < 8; ++j) rm = fmaxf(rm, mv[j]);
    rm = fmaxf(rm, __shfl_xor(rm, 16, 32));

    const float mnew = fmaxf(mstate, rm);
    const float corr = exp2f(mstate - mnew);
#pragma unroll
    for (int t = 0; t < 4; ++t)
#pragma unroll
      for (int r = 0; r < 8; ++r) st[t][r] = exp2f(st[t][r] - mnew);

    v8f sv = (st[0] + st[1]) + (st[2] + st[3]);
    float rs = sv[0];
#pragma unroll
    for (int j = 1; j < 8; ++j) rs += sv[j];
    rs += __shfl_xor(rs, 16, 32);

    lstate = lstate * corr + rs;
    mstate = mnew;
#pragma unroll
    for (int td = 0; td < 4; ++td) o[td] *= corr;

    // ---- P^T -> LDS: keys are contiguous per lane -> 4 packed b128 stores
#pragma unroll
    for (int t = 0; t < 4; ++t) {
      v8bf pk;
#pragma unroll
      for (int r = 0; r < 8; ++r) pk[r] = (__bf16)st[t][r];
      *(v8bf*)(&pbuf[wave][n][16 * t + 8 * h]) = pk;
    }

    // ---- reload P^T as B-fragments (wave-private region, LDS in-order)
    F16 bp[2];
    {
      const __bf16* prow = &pbuf[wave][n][0];
#pragma unroll
      for (int kk = 0; kk < 2; ++kk) {
        bp[kk].h[0] = *(const v8bf*)(prow + 32 * kk + 8 * h);
        bp[kk].h[1] = *(const v8bf*)(prow + 32 * kk + 16 + 8 * h);
      }
    }

    // ---- O^T += V^T @ P^T : 4 d-subtiles (A) x 2 key slabs -> 8 WMMA
#pragma unroll
    for (int td = 0; td < 4; ++td) {
      const __bf16* vrow = vT + (size_t)(16 * td + n) * SEQ + kb;
#pragma unroll
      for (int kk = 0; kk < 2; ++kk) {
        F16 av_;
        av_.h[0] = *(const v8bf*)(vrow + 32 * kk + 8 * h);
        av_.h[1] = *(const v8bf*)(vrow + 32 * kk + 16 + 8 * h);
        o[td] = __builtin_amdgcn_wmma_f32_16x16x32_bf16(
            false, av_.v, false, bp[kk].v, (short)0, o[td], false, false);
      }
    }
  }

  // ---- epilogue: per-lane 1/l, 8 consecutive d per (td,h) -> b128 stores
  const float inv = 1.0f / lstate;
  float* orow = out + (size_t)qrow * DM;
#pragma unroll
  for (int td = 0; td < 4; ++td) {
    F8F u; u.v = o[td] * inv;
    *(v4f*)(orow + 16 * td + 8 * h)     = u.q[0];
    *(v4f*)(orow + 16 * td + 8 * h + 4) = u.q[1];
  }
}

// ---------------------------------------------------------------------------
extern "C" void kernel_launch(void* const* d_in, const int* in_sizes, int n_in,
                              void* d_out, int out_size, void* d_ws, size_t ws_size,
                              hipStream_t stream) {
  const float* x  = (const float*)d_in[0];
  const float* wq = (const float*)d_in[1];
  const float* bq = (const float*)d_in[2];
  const float* wk = (const float*)d_in[3];
  const float* bk = (const float*)d_in[4];
  const float* wv = (const float*)d_in[5];
  const float* bv = (const float*)d_in[6];
  float* out = (float*)d_out;

  // workspace: Q (1MB) | K (1MB) | V^T (1MB), all bf16 -> resident in L2
  __bf16* q  = (__bf16*)d_ws;
  __bf16* k  = q + (size_t)SEQ * DM;
  __bf16* vT = k + (size_t)SEQ * DM;

  qkv_project<<<SEQ / 4, 256, 0, stream>>>(x, wq, bq, wk, bk, wv, bv, q, k, vT);
  flash_attn<<<SEQ / 128, 256, 0, stream>>>(q, k, vT, out);
}
